// MambaLikeBlock4D_86294482911296
// MI455X (gfx1250) — matrix operand order repaired and checked by reading, not compiled, not run on hardware
//
#include <hip/hip_runtime.h>

typedef __attribute__((ext_vector_type(16))) __bf16 v16bf;
typedef __attribute__((ext_vector_type(8)))  __bf16 v8bf;
typedef __attribute__((ext_vector_type(8)))  float  v8f;

#define B_DIM 8
#define C_DIM 1024
#define T_DIM 3000
#define T_PAD 3008
#define K_DIM 1024
#define M_TILE 128
#define N_TILE 64
#define K_TILE 32
#define LDS_STRIDE 40   // elements; 80B row stride -> 16B aligned, bank stride 20 (conflict-free)
#define SSM_K 24
#define TCH 512
#define HALO 23

// ---------------------------------------------------------------- stats (2-pass, deterministic)
__global__ __launch_bounds__(256) void mlb_stats_partial(const float* __restrict__ x,
                                                         float* __restrict__ partials) {
  __shared__ float ssum[256], ssq[256];
  const int b = blockIdx.y;
  const float* xb = x + (size_t)b * (C_DIM * T_DIM);
  float s = 0.f, q = 0.f;
  for (int i = blockIdx.x * 256 + threadIdx.x; i < C_DIM * T_DIM; i += 64 * 256) {
    float v = xb[i]; s += v; q += v * v;
  }
  ssum[threadIdx.x] = s; ssq[threadIdx.x] = q;
  __syncthreads();
  for (int st = 128; st > 0; st >>= 1) {
    if (threadIdx.x < st) { ssum[threadIdx.x] += ssum[threadIdx.x + st];
                            ssq [threadIdx.x] += ssq [threadIdx.x + st]; }
    __syncthreads();
  }
  if (threadIdx.x == 0) {
    partials[(b * 64 + blockIdx.x) * 2 + 0] = ssum[0];
    partials[(b * 64 + blockIdx.x) * 2 + 1] = ssq[0];
  }
}

__global__ __launch_bounds__(64) void mlb_stats_final(const float* __restrict__ partials,
                                                      float* __restrict__ stats) {
  __shared__ float ssum[64], ssq[64];
  const int b = blockIdx.x, t = threadIdx.x;
  ssum[t] = partials[(b * 64 + t) * 2 + 0];
  ssq [t] = partials[(b * 64 + t) * 2 + 1];
  __syncthreads();
  for (int st = 32; st > 0; st >>= 1) {
    if (t < st) { ssum[t] += ssum[t + st]; ssq[t] += ssq[t + st]; }
    __syncthreads();
  }
  if (t == 0) {
    const float inv = 1.f / (float)(C_DIM * T_DIM);
    float mu  = ssum[0] * inv;
    float var = ssq[0] * inv - mu * mu;
    stats[b] = mu;
    stats[B_DIM + b] = rsqrtf(var + 1e-6f);
  }
}

// ---------------------------------------------------------------- weight fp32 -> bf16
__global__ __launch_bounds__(256) void mlb_f32_to_bf16(const float* __restrict__ src,
                                                       __bf16* __restrict__ dst, int n) {
  int i = blockIdx.x * 256 + threadIdx.x;
  if (i < n) dst[i] = (__bf16)src[i];
}

// ---------------------------------------------------------------- groupnorm + transpose -> Hn[b][t][c] bf16
__global__ __launch_bounds__(256) void mlb_norm_transpose(const float* __restrict__ x,
                                                          const float* __restrict__ gamma,
                                                          const float* __restrict__ beta,
                                                          const float* __restrict__ stats,
                                                          __bf16* __restrict__ Hn) {
  __shared__ float tile[32][33];
  const int b = blockIdx.z;
  const int t0 = blockIdx.x * 32;
  const int c0 = blockIdx.y * 32;
  const int tx = threadIdx.x, ty = threadIdx.y;
  const float mu = stats[b], rs = stats[B_DIM + b];
  const int t = t0 + tx;
#pragma unroll
  for (int i = 0; i < 4; ++i) {
    const int c = c0 + ty + i * 8;
    float v = 0.f;
    if (t < T_DIM) {
      float xv = x[((size_t)b * C_DIM + c) * T_DIM + t];
      v = (xv - mu) * rs * gamma[c] + beta[c];
    }
    tile[tx][ty + i * 8] = v;     // [t_local][c_local]
  }
  __syncthreads();
#pragma unroll
  for (int i = 0; i < 4; ++i) {
    const int tt = t0 + ty + i * 8;
    const int c  = c0 + tx;
    Hn[((size_t)b * T_PAD + tt) * C_DIM + c] = (__bf16)tile[ty + i * 8][tx];
  }
}

// ---------------------------------------------------------------- WMMA GEMM: Out[o,t] = W[o,:] . Bt[t,:] + bias (+resid)
// Double-buffered LDS staged via GLOBAL_LOAD_ASYNC_TO_LDS_B128 (ASYNCcnt), one barrier per K-step.
// Staging addresses are loop-carried and kept opaque (asm "+v") so LLVM cannot strength-reduce them
// into dozens of hoisted 64-bit address pairs (VGPR pressure).
// MODE 0: write h2 fp32 [b][Mtot][T_PAD].  MODE 2: write d_out [b][Mtot][T_DIM] with residual xs.
template <int MODE>
__global__ __launch_bounds__(256) void mlb_gemm_wmma(const __bf16* __restrict__ Wb,
                                                     const __bf16* __restrict__ Bt,
                                                     const float* __restrict__ bias,
                                                     const float* __restrict__ resid,
                                                     float* __restrict__ Out,
                                                     int out_row_stride, int out_batch_stride) {
  __shared__ __bf16 As[2 * M_TILE * LDS_STRIDE];
  __shared__ __bf16 Bs[2 * N_TILE * LDS_STRIDE];
  const int b    = blockIdx.z;
  const int M0   = blockIdx.y * M_TILE;
  const int N0   = blockIdx.x * N_TILE;
  const int tid  = threadIdx.x;
  const int lane = tid & 31;
  const int wave = tid >> 5;
  const int mw   = (wave >> 1) * 32;   // 0,32,64,96
  const int nw   = (wave & 1) * 32;    // 0,32
  const __bf16* Btb = Bt + (size_t)b * T_PAD * K_DIM;

  const int arow = tid >> 1, ahalf = (tid & 1) * 16;  // A: 128 rows x 32 K, 32B/thread
  const int brow = tid >> 2, bq    = (tid & 3) * 8;   // B:  64 rows x 32 K, 16B/thread
  const int lhalf = lane >> 4, l15 = lane & 15;

  const __bf16* aSrc = Wb  + (size_t)(M0 + arow) * K_DIM + ahalf;
  const __bf16* bSrc = Btb + (size_t)(N0 + brow) * K_DIM + bq;
  // raw LDS byte offsets (low 32 bits of generic address == LDS offset per flat aperture rules)
  const uint32_t aD0 = (uint32_t)(size_t)&As[arow * LDS_STRIDE + ahalf];
  const uint32_t aD1 = aD0 + (uint32_t)(M_TILE * LDS_STRIDE * sizeof(__bf16));
  const uint32_t bD0 = (uint32_t)(size_t)&Bs[brow * LDS_STRIDE + bq];
  const uint32_t bD1 = bD0 + (uint32_t)(N_TILE * LDS_STRIDE * sizeof(__bf16));

  v8f acc[2][2];
#pragma unroll
  for (int m = 0; m < 2; ++m)
#pragma unroll
    for (int n = 0; n < 2; ++n) acc[m][n] = (v8f){0.f,0.f,0.f,0.f,0.f,0.f,0.f,0.f};

  // stage K-tile 0 into buffer 0 (inst offset applies to BOTH LDS and global address)
  {
    const uint64_t g0a = (uint64_t)(size_t)aSrc;
    const uint64_t g0b = (uint64_t)(size_t)bSrc;
    asm volatile("global_load_async_to_lds_b128 %0, %1, off"           :: "v"(aD0), "v"(g0a) : "memory");
    asm volatile("global_load_async_to_lds_b128 %0, %1, off offset:16" :: "v"(aD0), "v"(g0a) : "memory");
    asm volatile("global_load_async_to_lds_b128 %0, %1, off"           :: "v"(bD0), "v"(g0b) : "memory");
  }
  asm volatile("s_wait_asynccnt 0" ::: "memory");
  __syncthreads();

  uint64_t ga = (uint64_t)(size_t)(aSrc + K_TILE);   // next-tile global addresses
  uint64_t gb = (uint64_t)(size_t)(bSrc + K_TILE);
  uint32_t adNxt = aD1, bdNxt = bD1;                 // next-tile LDS destinations
  int aOffR = 0, bOffR = 0;                          // current read-buffer element offsets

#pragma clang loop unroll_count(2)
  for (int k0 = 0; k0 < K_DIM; k0 += K_TILE) {
    const bool more = (k0 + K_TILE) < K_DIM;
    if (more) {
      asm volatile("global_load_async_to_lds_b128 %0, %1, off"           :: "v"(adNxt), "v"(ga) : "memory");
      asm volatile("global_load_async_to_lds_b128 %0, %1, off offset:16" :: "v"(adNxt), "v"(ga) : "memory");
      asm volatile("global_load_async_to_lds_b128 %0, %1, off"           :: "v"(bdNxt), "v"(gb) : "memory");
      __builtin_prefetch((const void*)(size_t)(ga + 2 * K_TILE), 0, 3);  // tile k+2 -> GL2
      __builtin_prefetch((const void*)(size_t)(gb + 2 * K_TILE), 0, 3);
      ga += 2 * K_TILE;    // K_TILE bf16 = 64 bytes
      gb += 2 * K_TILE;
      asm volatile("" : "+v"(ga), "+v"(gb));         // keep loop-carried; block IV rewriting
    }

    union { v16bf v; v8bf h[2]; } afrag[2], bfrag[2];
#pragma unroll
    for (int m = 0; m < 2; ++m) {
      const __bf16* base = &As[aOffR + (mw + m * 16 + l15) * LDS_STRIDE];
      afrag[m].h[0] = *(const v8bf*)(base + lhalf * 8);        // K = khalf+0..7
      afrag[m].h[1] = *(const v8bf*)(base + 16 + lhalf * 8);   // K = 16+khalf+0..7
    }
#pragma unroll
    for (int n = 0; n < 2; ++n) {
      const __bf16* base = &Bs[bOffR + (nw + n * 16 + l15) * LDS_STRIDE];
      bfrag[n].h[0] = *(const v8bf*)(base + lhalf * 8);        // K = khalf+0..7
      bfrag[n].h[1] = *(const v8bf*)(base + 16 + lhalf * 8);   // K = 16+khalf+0..7
    }
#pragma unroll
    for (int m = 0; m < 2; ++m)
#pragma unroll
      for (int n = 0; n < 2; ++n)
        acc[m][n] = __builtin_amdgcn_wmma_f32_16x16x32_bf16(
            false, afrag[m].v, false, bfrag[n].v, (short)0, acc[m][n], false, false);

    if (more) {
      asm volatile("s_wait_asynccnt 0" ::: "memory");  // next buffer landed (this wave's DMAs)
      __syncthreads();                                 // ...and everyone else's; cur buffer fully read
      aOffR = (M_TILE * LDS_STRIDE) - aOffR;           // toggle read buffers
      bOffR = (N_TILE * LDS_STRIDE) - bOffR;
      adNxt = (aD0 + aD1) - adNxt;                     // toggle write destinations
      bdNxt = (bD0 + bD1) - bdNxt;
    }
  }

  // epilogue: C/D layout — VGPR v: M = v + 8*(lane>>4), N = lane&15
#pragma unroll
  for (int m = 0; m < 2; ++m) {
#pragma unroll
    for (int n = 0; n < 2; ++n) {
      const int Nabs = N0 + nw + n * 16 + l15;
#pragma unroll
      for (int v = 0; v < 8; ++v) {
        const int Mabs = M0 + mw + m * 16 + v + 8 * lhalf;
        float val = acc[m][n][v] + bias[Mabs];
        const size_t oidx = (size_t)b * out_batch_stride + (size_t)Mabs * out_row_stride + Nabs;
        if (MODE == 0) {
          Out[oidx] = val;                       // padded t columns stored, never consumed
        } else {
          if (Nabs < T_DIM) Out[oidx] = val + resid[oidx];
        }
      }
    }
  }
}

// ---------------------------------------------------------------- fused dconv + silu + SSM conv + gate -> Hsel[b][t][c] bf16
__global__ __launch_bounds__(256) void mlb_conv_gate_ssm(const float* __restrict__ h2,
                                                         const float* __restrict__ x,
                                                         const float* __restrict__ w_dconv,
                                                         const float* __restrict__ b_dconv,
                                                         const float* __restrict__ alpha,
                                                         const float* __restrict__ beta,
                                                         const float* __restrict__ theta,
                                                         __bf16* __restrict__ Hsel) {
  __shared__ float hc[TCH + HALO];
  __shared__ float ker[SSM_K];
  const int b = blockIdx.z, c = blockIdx.y, t0 = blockIdx.x * TCH;
  const int tid = threadIdx.x;
  if (tid < SSM_K) {
    float la  = logf(fmaxf(alpha[c], 1e-6f));
    float j   = (float)tid;
    float dec = expf(la * j);
    float z   = theta[c] * j;
    float z2  = z * z;
    float ph  = 1.f - 0.5f * z2 + z2 * z2 * (1.f / 24.f);
    ker[tid]  = beta[c] * dec * ph;
  }
  const float* hcc = h2 + ((size_t)b * 2 * C_DIM + c) * T_PAD;
  const float* hgg = h2 + ((size_t)b * 2 * C_DIM + C_DIM + c) * T_PAD;
  const float w0 = w_dconv[c * 3 + 0], w1 = w_dconv[c * 3 + 1], w2 = w_dconv[c * 3 + 2];
  const float bb = b_dconv[c];
  for (int idx = tid; idx < TCH + HALO; idx += 256) {
    const int t = t0 - HALO + idx;
    float v = 0.f;
    if (t >= 0 && t < T_DIM) {
      float s = bb + w2 * hcc[t];
      if (t >= 1) s += w1 * hcc[t - 1];
      if (t >= 2) s += w0 * hcc[t - 2];
      v = s / (1.f + expf(-s));                 // silu
    }
    hc[idx] = v;
  }
  __syncthreads();
  const float gw0 = w_dconv[(C_DIM + c) * 3 + 0];
  const float gw1 = w_dconv[(C_DIM + c) * 3 + 1];
  const float gw2 = w_dconv[(C_DIM + c) * 3 + 2];
  const float gbb = b_dconv[C_DIM + c];
  for (int idx = tid; idx < TCH; idx += 256) {
    const int t = t0 + idx;
    if (t < T_DIM) {
      float ssm = 0.f;
#pragma unroll
      for (int j = 0; j < SSM_K; ++j) ssm += ker[j] * hc[idx + j];   // out[t]=sum_j k[j]*in[t-23+j]
      float g = gbb + gw2 * hgg[t];
      if (t >= 1) g += gw1 * hgg[t - 1];
      if (t >= 2) g += gw0 * hgg[t - 2];
      float gate = 1.f / (1.f + expf(-g));
      float xv = x[((size_t)b * C_DIM + c) * T_DIM + t];
      float hsel = (ssm + xv * 1e-9f) * (gate + xv * 1e-9f);
      Hsel[((size_t)b * T_PAD + t) * C_DIM + c] = (__bf16)hsel;
    }
  }
}

// ---------------------------------------------------------------- launcher
extern "C" void kernel_launch(void* const* d_in, const int* in_sizes, int n_in,
                              void* d_out, int out_size, void* d_ws, size_t ws_size,
                              hipStream_t stream) {
  const float* x        = (const float*)d_in[0];
  const float* gn_gamma = (const float*)d_in[1];
  const float* gn_beta  = (const float*)d_in[2];
  const float* w_in     = (const float*)d_in[3];
  const float* b_in     = (const float*)d_in[4];
  const float* w_dconv  = (const float*)d_in[5];
  const float* b_dconv  = (const float*)d_in[6];
  const float* alpha    = (const float*)d_in[7];
  const float* beta     = (const float*)d_in[8];
  const float* theta    = (const float*)d_in[9];
  const float* w_out    = (const float*)d_in[10];
  const float* b_out    = (const float*)d_in[11];
  float* out = (float*)d_out;

  char* wsb = (char*)d_ws;
  float*  partials = (float*)wsb;                               // 4096 B
  float*  stats    = (float*)(wsb + 4096);                      // 64 B (pad to 8192)
  __bf16* W1b      = (__bf16*)(wsb + 8192);                     // 2048*1024 bf16
  __bf16* W2b      = W1b + (size_t)2048 * 1024;                 // 1024*1024 bf16
  __bf16* Hn       = W2b + (size_t)1024 * 1024;                 // 8*3008*1024 bf16
  __bf16* Hsel     = Hn  + (size_t)B_DIM * T_PAD * C_DIM;       // 8*3008*1024 bf16
  float*  h2       = (float*)(Hsel + (size_t)B_DIM * T_PAD * C_DIM); // 8*2048*3008 f32

  mlb_stats_partial<<<dim3(64, B_DIM), 256, 0, stream>>>(x, partials);
  mlb_stats_final<<<B_DIM, 64, 0, stream>>>(partials, stats);
  mlb_f32_to_bf16<<<(2048 * 1024 + 255) / 256, 256, 0, stream>>>(w_in, W1b, 2048 * 1024);
  mlb_f32_to_bf16<<<(1024 * 1024 + 255) / 256, 256, 0, stream>>>(w_out, W2b, 1024 * 1024);
  mlb_norm_transpose<<<dim3(T_PAD / 32, C_DIM / 32, B_DIM), dim3(32, 8), 0, stream>>>(
      x, gn_gamma, gn_beta, stats, Hn);
  mlb_gemm_wmma<0><<<dim3(T_PAD / N_TILE, 2048 / M_TILE, B_DIM), 256, 0, stream>>>(
      W1b, Hn, b_in, nullptr, h2, T_PAD, 2048 * T_PAD);
  mlb_conv_gate_ssm<<<dim3((T_DIM + TCH - 1) / TCH, C_DIM, B_DIM), 256, 0, stream>>>(
      h2, x, w_dconv, b_dconv, alpha, beta, theta, Hsel);
  mlb_gemm_wmma<2><<<dim3(T_PAD / N_TILE, 1024 / M_TILE, B_DIM), 256, 0, stream>>>(
      W2b, Hsel, b_out, x, out, T_DIM, 1024 * T_DIM);
}
